// ConvVFE_44092134261311
// MI455X (gfx1250) — matrix operand order repaired and checked by reading, not compiled, not run on hardware
//
#include <hip/hip_runtime.h>
#include <hip/hip_bf16.h>

typedef __attribute__((ext_vector_type(16))) _Float16 v16h;
typedef __attribute__((ext_vector_type(8)))  _Float16 v8h;
typedef __attribute__((ext_vector_type(8)))  float    v8f;

#define NKEYS 220000
#define SXYZ_ 55000
#define SYZ_  250
#define GX_ 220
#define GY_ 250

// ---------------- feats helper ----------------
__device__ __forceinline__ void vfe_feats(float x, float y, float z, float it,
                                          float mx, float my, float mz,
                                          int cx, int cy, float f[10]) {
  f[0] = x; f[1] = y; f[2] = z; f[3] = it;
  f[4] = x - mx; f[5] = y - my; f[6] = z - mz;
  f[7] = x - ((float)cx * 0.32f + 0.16f);
  f[8] = y - ((float)cy * 0.32f - 39.84f);
  f[9] = z + 1.0f;
}

// ---------------- zero ----------------
__global__ void kzero(unsigned* __restrict__ p, long n) {
  long stride = (long)gridDim.x * blockDim.x;
  for (long i = (long)blockIdx.x * blockDim.x + threadIdx.x; i < n; i += stride) p[i] = 0u;
}

// ---------------- K1: keys, counts, xyz sums ----------------
__global__ void k1_keys(const float* __restrict__ pts, int N, int* __restrict__ keys,
                        unsigned* __restrict__ cnt, float* __restrict__ sx,
                        float* __restrict__ sy, float* __restrict__ sz,
                        unsigned* __restrict__ meta) {
  __shared__ unsigned nloc;
  if (threadIdx.x == 0) nloc = 0u;
  __syncthreads();
  int stride = gridDim.x * blockDim.x;
  for (int i = blockIdx.x * blockDim.x + threadIdx.x; i < N; i += stride) {
    float b = pts[i * 5 + 0], x = pts[i * 5 + 1], y = pts[i * 5 + 2], z = pts[i * 5 + 3];
    int cx = (int)floorf(x / 0.32f);
    int cy = (int)floorf((y + 40.0f) / 0.32f);
    int cz = (int)floorf((z + 3.0f) / 4.0f);
    int key = -1;
    if (cx >= 0 && cx < GX_ && cy >= 0 && cy < GY_ && cz >= 0 && cz < 1) {
      key = (int)b * SXYZ_ + cx * SYZ_ + cy + cz;
      atomicAdd(&cnt[key], 1u);
      atomicAdd(&sx[key], x);
      atomicAdd(&sy[key], y);
      atomicAdd(&sz[key], z);
      atomicAdd(&nloc, 1u);
    }
    keys[i] = key;
  }
  __syncthreads();
  if (threadIdx.x == 0) atomicAdd(&meta[1], nloc);
}

// ---------------- K2a: exclusive scan over dense keys (unique) ----------------
__global__ void k2a_scan(const unsigned* __restrict__ cnt, int* __restrict__ voxid,
                         unsigned* __restrict__ meta) {
  __shared__ unsigned temp[1024];
  __shared__ unsigned carry;
  int tid = threadIdx.x;
  if (tid == 0) carry = 0u;
  __syncthreads();
  for (int base = 0; base < NKEYS; base += 1024) {
    int i = base + tid;
    unsigned f = (i < NKEYS && cnt[i] > 0u) ? 1u : 0u;
    temp[tid] = f;
    __syncthreads();
    for (int off = 1; off < 1024; off <<= 1) {
      unsigned v = temp[tid];
      unsigned add = (tid >= off) ? temp[tid - off] : 0u;
      __syncthreads();
      temp[tid] = v + add;
      __syncthreads();
    }
    unsigned excl = temp[tid] - f;
    if (i < NKEYS) voxid[i] = f ? (int)(carry + excl) : -1;
    __syncthreads();
    if (tid == 0) carry += temp[1023];
    __syncthreads();
  }
  if (tid == 0) meta[0] = carry;  // V
}

// ---------------- K2b: pillar_coords ----------------
__global__ void k2b_coords(const int* __restrict__ voxid, const unsigned* __restrict__ meta,
                           int* __restrict__ outI) {
  long base = (long)meta[0] * 128;
  int stride = gridDim.x * blockDim.x;
  for (int k = blockIdx.x * blockDim.x + threadIdx.x; k < NKEYS; k += stride) {
    int v = voxid[k];
    if (v >= 0) {
      outI[base + (long)v * 4 + 0] = k / SXYZ_;
      outI[base + (long)v * 4 + 1] = 0;
      outI[base + (long)v * 4 + 2] = k % 250;
      outI[base + (long)v * 4 + 3] = (k % SXYZ_) / 250;
    }
  }
}

// ---------------- layer-0 per-point feats ----------
__device__ __forceinline__ void layer0_eval(const float* __restrict__ pts, int pt, int key,
                                            const unsigned* __restrict__ cnt,
                                            const float* __restrict__ sx,
                                            const float* __restrict__ sy,
                                            const float* __restrict__ sz,
                                            float f[10]) {
  float x = pts[pt * 5 + 1], y = pts[pt * 5 + 2], z = pts[pt * 5 + 3], it = pts[pt * 5 + 4];
  float c = (float)cnt[key];
  float mx = sx[key] / c, my = sy[key] / c, mz = sz[key] / c;
  int rem = key % SXYZ_;
  int cx = rem / 250, cy = rem % 250;
  vfe_feats(x, y, z, it, mx, my, mz, cx, cy, f);
}

// ---------------- K3: BN0 statistics ----------------
__global__ void k3_bn0stats(const float* __restrict__ pts, int N, const int* __restrict__ keys,
                            const unsigned* __restrict__ cnt, const float* __restrict__ sx,
                            const float* __restrict__ sy, const float* __restrict__ sz,
                            const float* __restrict__ w0, float* __restrict__ bn0sum,
                            float* __restrict__ bn0sq) {
  __shared__ float w0L[320];
  __shared__ float sumL[32], sqL[32];
  for (int i = threadIdx.x; i < 320; i += blockDim.x) w0L[i] = w0[i];
  if (threadIdx.x < 32) { sumL[threadIdx.x] = 0.f; sqL[threadIdx.x] = 0.f; }
  __syncthreads();
  float ls[32], lq[32];
#pragma unroll
  for (int c = 0; c < 32; c++) { ls[c] = 0.f; lq[c] = 0.f; }
  int stride = gridDim.x * blockDim.x;
  for (int i = blockIdx.x * blockDim.x + threadIdx.x; i < N; i += stride) {
    int key = keys[i];
    if (key < 0) continue;
    float f[10];
    layer0_eval(pts, i, key, cnt, sx, sy, sz, f);
#pragma unroll
    for (int c = 0; c < 32; c++) {
      float acc = 0.f;
#pragma unroll
      for (int k = 0; k < 10; k++) acc = fmaf(f[k], w0L[k * 32 + c], acc);
      ls[c] += acc;
      lq[c] += acc * acc;
    }
  }
#pragma unroll
  for (int c = 0; c < 32; c++) { atomicAdd(&sumL[c], ls[c]); atomicAdd(&sqL[c], lq[c]); }
  __syncthreads();
  if (threadIdx.x < 32) {
    atomicAdd(&bn0sum[threadIdx.x], sumL[threadIdx.x]);
    atomicAdd(&bn0sq[threadIdx.x], sqL[threadIdx.x]);
  }
}

// ---------------- K4: BN0+ReLU, per-voxel max of 32-dim ----------------
__global__ void k4_segmax0(const float* __restrict__ pts, int N, const int* __restrict__ keys,
                           const int* __restrict__ voxid, const unsigned* __restrict__ cnt,
                           const float* __restrict__ sx, const float* __restrict__ sy,
                           const float* __restrict__ sz, const float* __restrict__ w0,
                           const float* __restrict__ bn0sum, const float* __restrict__ bn0sq,
                           const unsigned* __restrict__ meta, const float* __restrict__ g0,
                           const float* __restrict__ b0, float* __restrict__ segmax0) {
  __shared__ float w0L[320];
  __shared__ float sc[32], sh[32];
  for (int i = threadIdx.x; i < 320; i += blockDim.x) w0L[i] = w0[i];
  if (threadIdx.x < 32) {
    float Pn = (float)meta[1]; if (Pn < 1.f) Pn = 1.f;
    float m = bn0sum[threadIdx.x] / Pn;
    float v = bn0sq[threadIdx.x] / Pn - m * m;
    float s = g0[threadIdx.x] * rsqrtf(v + 1e-3f);
    sc[threadIdx.x] = s; sh[threadIdx.x] = b0[threadIdx.x] - m * s;
  }
  __syncthreads();
  int stride = gridDim.x * blockDim.x;
  for (int i = blockIdx.x * blockDim.x + threadIdx.x; i < N; i += stride) {
    int key = keys[i];
    if (key < 0) continue;
    int vox = voxid[key];
    float f[10];
    layer0_eval(pts, i, key, cnt, sx, sy, sz, f);
#pragma unroll
    for (int c = 0; c < 32; c++) {
      float acc = 0.f;
#pragma unroll
      for (int k = 0; k < 10; k++) acc = fmaf(f[k], w0L[k * 32 + c], acc);
      float a = fmaxf(0.f, fmaf(acc, sc[c], sh[c]));
      atomicMax((int*)&segmax0[(long)vox * 32 + c], __float_as_int(a));
    }
  }
}

// ---------------- K5/K6: WMMA layer-1 (64 -> 128) ----------------
// FINAL==0: accumulate BN1 sum/sumsq. Invalid rows are all-zero in x64, so their
//           WMMA output rows are exactly zero and need NO validity masking.
// FINAL==1: BN1+ReLU, segmax into out (validity check routes the atomic).
// B operand lives in LDS swizzled as [n][k] so fragments are vector ds loads.
template <int FINAL>
__global__ void k56_layer1(const float* __restrict__ pts, int N, const int* __restrict__ keys,
                           const int* __restrict__ voxid, const unsigned* __restrict__ cnt,
                           const float* __restrict__ sx, const float* __restrict__ sy,
                           const float* __restrict__ sz, const float* __restrict__ w0,
                           const float* __restrict__ w1, const float* __restrict__ bn0sum,
                           const float* __restrict__ bn0sq, const float* __restrict__ g0,
                           const float* __restrict__ b0, float* __restrict__ bn1sum,
                           float* __restrict__ bn1sq, const float* __restrict__ g1,
                           const float* __restrict__ b1, const unsigned* __restrict__ meta,
                           const float* __restrict__ segmax0, float* __restrict__ out) {
  __shared__ _Float16 w1hT[128 * 64];     // transposed weights: [n][k], rows of 64 halves
  __shared__ _Float16 x64[8][16][64];     // per-wave 16-point x 64-feature tiles
  __shared__ int vox8[8][16];
  __shared__ float w0L[320];
  __shared__ float sc0[32], sh0[32];
  __shared__ float bs[128], bq[128];

  const int tid = threadIdx.x;
  // fill transposed f16 weights with coalesced global reads:
  // w1hT[n*64 + k] = w1[k*128 + n]
  for (int i = tid; i < 128 * 64; i += blockDim.x) {
    int kk = i >> 7, nn = i & 127;
    w1hT[nn * 64 + kk] = (_Float16)w1[i];
  }
  for (int i = tid; i < 320; i += blockDim.x) w0L[i] = w0[i];
  float Pn = (float)meta[1]; if (Pn < 1.f) Pn = 1.f;
  if (tid < 32) {
    float m = bn0sum[tid] / Pn;
    float v = bn0sq[tid] / Pn - m * m;
    float s = g0[tid] * rsqrtf(v + 1e-3f);
    sc0[tid] = s; sh0[tid] = b0[tid] - m * s;
  }
  if (FINAL == 0 && tid < 128) { bs[tid] = 0.f; bq[tid] = 0.f; }

  const int lane = tid & 31, w = tid >> 5;
  const int p = lane & 15, g = lane >> 4;   // point-in-tile, half-index
  const int n = lane & 15;                  // output column within 16-wide tile
  const int kb = g * 16;                    // unused; B fragments start at K = g*8 and 16+g*8

  float sc1[8], sh1[8];
  if (FINAL) {
#pragma unroll
    for (int nt = 0; nt < 8; nt++) {
      int ch = nt * 16 + n;
      float m = bn1sum[ch] / Pn;
      float v = bn1sq[ch] / Pn - m * m;
      float s = g1[ch] * rsqrtf(v + 1e-3f);
      sc1[nt] = s; sh1[nt] = b1[ch] - m * s;
    }
  }
  __syncthreads();

  float sAcc[8], qAcc[8];
#pragma unroll
  for (int nt = 0; nt < 8; nt++) { sAcc[nt] = 0.f; qAcc[nt] = 0.f; }

  const int ntiles = (N + 15) >> 4;
  for (int t0 = blockIdx.x * 8; t0 < ntiles; t0 += gridDim.x * 8) {
    int tile = t0 + w;
    if (tile < ntiles) {
      int pt = tile * 16 + p;
      int key = (pt < N) ? keys[pt] : -1;
      int c0 = g * 16;
      v16h av, sv;
      int vx = -1;
      if (key >= 0) {
        vx = voxid[key];
        float f[10];
        layer0_eval(pts, pt, key, cnt, sx, sy, sz, f);
#pragma unroll
        for (int j = 0; j < 16; j++) {
          int ch = c0 + j;
          float acc = 0.f;
#pragma unroll
          for (int k = 0; k < 10; k++) acc = fmaf(f[k], w0L[k * 32 + ch], acc);
          av[j] = (_Float16)fmaxf(0.f, fmaf(acc, sc0[ch], sh0[ch]));
          sv[j] = (_Float16)segmax0[(long)vx * 32 + ch];
        }
      } else {
#pragma unroll
        for (int j = 0; j < 16; j++) { av[j] = (_Float16)0.f; sv[j] = (_Float16)0.f; }
      }
      *(v16h*)&x64[w][p][c0] = av;        // a-part
      *(v16h*)&x64[w][p][32 + c0] = sv;   // segmax-part
      if (FINAL && g == 0) vox8[w][p] = vx;
    }
    __syncthreads();
    if (tile < ntiles) {
      // A fragments (16-bit A 16x32): two contiguous 8-half runs per half-K
      const _Float16* ap = &x64[w][p][0];
      v8h a0 = *(const v8h*)(ap + g * 8);        // K  g*8 .. g*8+7
      v8h a1 = *(const v8h*)(ap + 16 + g * 8);   // K 16+g*8 ..
      v8h a2 = *(const v8h*)(ap + 32 + g * 8);   // K 32+g*8 ..
      v8h a3 = *(const v8h*)(ap + 48 + g * 8);   // K 48+g*8 ..
      v16h alo = __builtin_shufflevector(a0, a1, 0, 1, 2, 3, 4, 5, 6, 7,
                                         8, 9, 10, 11, 12, 13, 14, 15);
      v16h ahi = __builtin_shufflevector(a2, a3, 0, 1, 2, 3, 4, 5, 6, 7,
                                         8, 9, 10, 11, 12, 13, 14, 15);
      int vvr[8];
      if (FINAL) {
#pragma unroll
        for (int r = 0; r < 8; r++) vvr[r] = vox8[w][r + g * 8];
      }
#pragma unroll
      for (int nt = 0; nt < 8; nt++) {
        // B fragments from swizzled LDS: Blo elements are K=g*8..g*8+7, 16+g*8..16+g*8+7 at col nt*16+n
        const _Float16* bp = &w1hT[(nt * 16 + n) * 64 + g * 8];
        v16h blo = __builtin_shufflevector(*(const v8h*)bp, *(const v8h*)(bp + 16), 0, 1, 2, 3, 4, 5, 6, 7, 8, 9, 10, 11, 12, 13, 14, 15);          // K = g*8.., 16+g*8..
        v16h bhi = __builtin_shufflevector(*(const v8h*)(bp + 32), *(const v8h*)(bp + 48), 0, 1, 2, 3, 4, 5, 6, 7, 8, 9, 10, 11, 12, 13, 14, 15);   // K = 32+g*8.., 48+g*8..
        v8f c = {};
        c = __builtin_amdgcn_wmma_f32_16x16x32_f16(false, alo, false, blo, (short)0, c,
                                                   false, false);
        c = __builtin_amdgcn_wmma_f32_16x16x32_f16(false, ahi, false, bhi, (short)0, c,
                                                   false, false);
        if (FINAL == 0) {
          // invalid rows are exactly zero: no masking needed
#pragma unroll
          for (int r = 0; r < 8; r++) {
            sAcc[nt] += c[r];
            qAcc[nt] = fmaf(c[r], c[r], qAcc[nt]);
          }
        } else {
#pragma unroll
          for (int r = 0; r < 8; r++) {
            if (vvr[r] >= 0) {
              float val = fmaxf(0.f, fmaf(c[r], sc1[nt], sh1[nt]));
              atomicMax((int*)&out[(long)vvr[r] * 128 + nt * 16 + n], __float_as_int(val));
            }
          }
        }
      }
    }
    __syncthreads();
  }
  if (FINAL == 0) {
#pragma unroll
    for (int nt = 0; nt < 8; nt++) {
      atomicAdd(&bs[nt * 16 + n], sAcc[nt]);
      atomicAdd(&bq[nt * 16 + n], qAcc[nt]);
    }
    __syncthreads();
    if (tid < 128) {
      atomicAdd(&bn1sum[tid], bs[tid]);
      atomicAdd(&bn1sq[tid], bq[tid]);
    }
  }
}

// ---------------- host-side launcher ----------------
extern "C" void kernel_launch(void* const* d_in, const int* in_sizes, int n_in,
                              void* d_out, int out_size, void* d_ws, size_t ws_size,
                              hipStream_t stream) {
  const float* pts = (const float*)d_in[0];
  const float* w0 = (const float*)d_in[1];
  const float* g0 = (const float*)d_in[2];
  const float* b0 = (const float*)d_in[3];
  const float* w1 = (const float*)d_in[4];
  const float* g1 = (const float*)d_in[5];
  const float* b1 = (const float*)d_in[6];
  const int N = in_sizes[0] / 5;

  // workspace carve-up (all 4-byte aligned)
  char* base = (char*)d_ws;
  unsigned* cnt = (unsigned*)base;                 // NKEYS
  float* sx = (float*)(cnt + NKEYS);               // NKEYS
  float* sy = sx + NKEYS;                          // NKEYS
  float* sz = sy + NKEYS;                          // NKEYS
  int* voxid = (int*)(sz + NKEYS);                 // NKEYS
  unsigned* meta = (unsigned*)(voxid + NKEYS);     // 8  ([0]=V, [1]=P_valid)
  float* bn0sum = (float*)(meta + 8);              // 32
  float* bn0sq = bn0sum + 32;                      // 32
  float* bn1sum = bn0sq + 32;                      // 128
  float* bn1sq = bn1sum + 128;                     // 128
  float* segmax0 = bn1sq + 128;                    // NKEYS*32
  int* keys = (int*)(segmax0 + (long)NKEYS * 32);  // N

  const long zero_u32 = 5L * NKEYS + 8 + 320 + 32L * NKEYS;  // through segmax0
  float* out = (float*)d_out;

  kzero<<<2048, 256, 0, stream>>>((unsigned*)d_ws, zero_u32);
  kzero<<<2048, 256, 0, stream>>>((unsigned*)d_out, (long)out_size);

  k1_keys<<<2048, 256, 0, stream>>>(pts, N, keys, cnt, sx, sy, sz, meta);
  k2a_scan<<<1, 1024, 0, stream>>>(cnt, voxid, meta);
  k2b_coords<<<512, 256, 0, stream>>>(voxid, meta, (int*)d_out);
  k3_bn0stats<<<2048, 256, 0, stream>>>(pts, N, keys, cnt, sx, sy, sz, w0, bn0sum, bn0sq);
  k4_segmax0<<<2048, 256, 0, stream>>>(pts, N, keys, voxid, cnt, sx, sy, sz, w0, bn0sum,
                                       bn0sq, meta, g0, b0, segmax0);
  k56_layer1<0><<<1024, 256, 0, stream>>>(pts, N, keys, voxid, cnt, sx, sy, sz, w0, w1,
                                          bn0sum, bn0sq, g0, b0, bn1sum, bn1sq, g1, b1,
                                          meta, segmax0, out);
  k56_layer1<1><<<1024, 256, 0, stream>>>(pts, N, keys, voxid, cnt, sx, sy, sz, w0, w1,
                                          bn0sum, bn0sq, g0, b0, bn1sum, bn1sq, g1, b1,
                                          meta, segmax0, out);
}
